// ResidualBlock_75505525063915
// MI455X (gfx1250) — matrix operand order repaired and checked by reading, not compiled, not run on hardware
//
#include <hip/hip_runtime.h>
#include <cmath>

#define PI_D 3.14159265358979323846
#define PI_F 3.14159265358979323846f

typedef __attribute__((ext_vector_type(16))) _Float16 v16h;
typedef __attribute__((ext_vector_type(8)))  float    v8f;

// ---------------------------------------------------------------------------
// Device math helpers (constants recomputed deterministically each launch)
// ---------------------------------------------------------------------------
__device__ inline double dev_fact(int n){ double f=1.0; for(int i=2;i<=n;++i) f*=(double)i; return f; }

// Quantum-convention Wigner small-d: d^l_{m'm}(beta) = <l m'| exp(-i beta Jy) |l m>
__device__ double dev_wigner_d(int l, int mp, int m, double beta){
  if (mp > l || mp < -l || m > l || m < -l) return 0.0;
  double ch = cos(0.5*beta), sh = sin(0.5*beta);
  double pref = sqrt(dev_fact(l+mp)*dev_fact(l-mp)*dev_fact(l+m)*dev_fact(l-m));
  int smin = (m - mp) > 0 ? (m - mp) : 0;
  int smax = (l + m) < (l - mp) ? (l + m) : (l - mp);
  double sum = 0.0;
  for (int s = smin; s <= smax; ++s){
    double den = dev_fact(l+m-s)*dev_fact(s)*dev_fact(mp-m+s)*dev_fact(l-mp-s);
    double t = pow(ch, (double)(2*l + m - mp - 2*s)) * pow(sh, (double)(mp - m + 2*s)) / den;
    if ((mp - m + s) & 1) t = -t;
    sum += t;
  }
  return pref * sum;
}

// Driscoll-Healy quadrature weight (input bandwidth b)
__device__ double dev_qw(int k, int b){
  double beta = PI_D * (2*k + 1) / (4.0*b);
  double s = 0.0;
  for (int j = 0; j < b; ++j)
    s += sin((2*k+1)*(2*j+1)*PI_D/(4.0*b)) / (double)(2*j+1);
  double cell = 2.0*PI_D/(2.0*b);
  return (2.0/b) * sin(beta) * s * cell * cell;
}

// ---------------------------------------------------------------------------
// Wf / Wi constants: out[k][l][p][q]; mode 0 => qw(k)*d, mode 1 => (2l+1)/(8pi^2)*d
// ---------------------------------------------------------------------------
__global__ void k_wig(float* out, int NB, int NL, int L, double bscale, int mode){
  long idx = (long)blockIdx.x*blockDim.x + threadIdx.x;
  long tot = (long)NB*NL*L*L;
  if (idx >= tot) return;
  int q = (int)(idx % L); long r = idx / L;
  int p = (int)(r % L);   r /= L;
  int l = (int)(r % NL);  int k = (int)(r / NL);
  int half = (L-1)/2;
  int mp = p - half, m = q - half;
  double beta = bscale * (2*k + 1);
  double d = dev_wigner_d(l, mp, m, beta);
  double w = (mode == 0) ? dev_qw(k, 16) : (2.0*l + 1.0)/(8.0*PI_D*PI_D);
  out[idx] = (float)(d * w);
}

// Dg[g][l][p][q] = d^l_{m'm}(pi/8) * exp(i*(alpha*m' + gamma*m)), grid g = ia*6 + ic
__global__ void k_dg(float2* out, int NL, int L){
  long idx = (long)blockIdx.x*blockDim.x + threadIdx.x;
  long tot = 36L*NL*L*L;
  if (idx >= tot) return;
  int q = (int)(idx % L); long r = idx / L;
  int p = (int)(r % L);   r /= L;
  int l = (int)(r % NL);  int g = (int)(r / NL);
  int half = (L-1)/2;
  int mp = p - half, m = q - half;
  double d = dev_wigner_d(l, mp, m, PI_D/8.0);
  int ia = g / 6, ic = g % 6;
  double alpha = ia * (2.0*PI_D/6.0);
  double gamma = (-2.0*PI_D + ic * (4.0*PI_D/6.0)) - alpha;
  double ph = alpha*mp + gamma*m;
  out[idx] = make_float2((float)(d*cos(ph)), (float)(d*sin(ph)));
}

// ---------------------------------------------------------------------------
// Forward DFT restricted to the L needed frequencies (m = p-(L-1)/2, period 32)
// in: [planes][32][32] real, out: [planes][L][L] complex.  One plane per block.
// ---------------------------------------------------------------------------
__global__ void k_dft(const float* __restrict__ in, float2* __restrict__ out, int L){
  __shared__ float  xs[1024];
  __shared__ float2 T[31*32];
  const int tid = threadIdx.x;
  const float* src = in + (size_t)blockIdx.x * 1024;
  for (int t = tid; t < 1024; t += 256) xs[t] = src[t];
  __syncthreads();
  const int half = (L-1)/2;
  for (int t = tid; t < L*32; t += 256){
    int p = t / 32, v = t % 32;
    int mf = p - half;
    float sr = 0.f, si = 0.f;
    for (int u = 0; u < 32; ++u){
      int ph = ((mf*u) % 32 + 32) % 32;
      float ang = -(2.0f*PI_F/32.0f) * (float)ph;
      float s, c; __sincosf(ang, &s, &c);
      float xv = xs[u*32 + v];
      sr += xv*c; si += xv*s;
    }
    T[p*32 + v] = make_float2(sr, si);
  }
  __syncthreads();
  float2* dst = out + (size_t)blockIdx.x * L * L;
  for (int t = tid; t < L*L; t += 256){
    int p = t / L, q = t % L;
    int nf = q - half;
    float sr = 0.f, si = 0.f;
    for (int v = 0; v < 32; ++v){
      int ph = ((nf*v) % 32 + 32) % 32;
      float ang = -(2.0f*PI_F/32.0f) * (float)ph;
      float s, c; __sincosf(ang, &s, &c);
      float2 tv = T[p*32 + v];
      sr += tv.x*c - tv.y*s;
      si += tv.x*s + tv.y*c;
    }
    dst[t] = make_float2(sr, si);
  }
}

// ---------------------------------------------------------------------------
// Fused beta-contraction + WMMA A-fragment pack.
// A fragment layout (CDNA5 ISA 7.12.2, 16x32 f16): m = lane&15,
//   k(lane,h) = (h<8 ? h : h+8) + 8*(lane>=16).
// Packed buffer: [l][mt][ch][lane][16 halves]; zero-filled padding.
// ---------------------------------------------------------------------------
__global__ void k_fh_pack(const float2* __restrict__ Fz, int Lsrc, int off,
                          const float* __restrict__ Wf,
                          _Float16* __restrict__ Apr, _Float16* __restrict__ Api,
                          int NL, int L, int MT, int NC){
  long idx = (long)blockIdx.x*blockDim.x + threadIdx.x;
  long tot = (long)NL*MT*NC*512;
  if (idx >= tot) return;
  int h    = (int)(idx & 15);
  int lane = (int)((idx >> 4) & 31);
  long r = idx >> 9;
  int ch = (int)(r % NC); r /= NC;
  int mt = (int)(r % MT); int l = (int)(r / MT);
  int kloc = (h < 8 ? h : h + 8) + ((lane >> 4) << 3);
  int kk   = ch*32 + kloc;
  int row  = mt*16 + (lane & 15);
  float sr = 0.f, si = 0.f;
  if (row < 8*L && kk < 16*L){
    int b = row / L, m = row % L;
    int i = kk / L,  k2 = kk % L;
    for (int kb = 0; kb < 32; ++kb){
      float wv = Wf[(((size_t)kb*NL + l)*L + m)*L + k2];
      float2 f = Fz[((((size_t)b*16 + i)*32 + kb)*Lsrc + (m + off))*Lsrc + (k2 + off)];
      sr += wv*f.x; si += wv*f.y;
    }
  }
  Apr[idx] = (_Float16)sr;
  Api[idx] = (_Float16)si;
}

// ---------------------------------------------------------------------------
// Fused Psi build + conj + WMMA B-fragment pack.
// B fragment layout (32x16 f16): n = lane&15, k = (h<8 ? h : h+8) + 8*(lane>=16).
// Packed buffer: [l][nt][ch][lane][16 halves]; zero-filled padding.
// ---------------------------------------------------------------------------
__global__ void k_psi_pack(const float* __restrict__ w, const float2* __restrict__ Dg,
                           _Float16* __restrict__ Bpr, _Float16* __restrict__ Bpi,
                           int CHO, int NL, int L, int NT, int NC, float scale){
  long idx = (long)blockIdx.x*blockDim.x + threadIdx.x;
  long tot = (long)NL*NT*NC*512;
  if (idx >= tot) return;
  int h    = (int)(idx & 15);
  int lane = (int)((idx >> 4) & 31);
  long r = idx >> 9;
  int ch = (int)(r % NC); r /= NC;
  int nt = (int)(r % NT); int l = (int)(r / NT);
  int kk  = ch*32 + (h < 8 ? h : h + 8) + ((lane >> 4) << 3);
  int col = nt*16 + (lane & 15);
  float ar = 0.f, ai = 0.f;
  if (kk < 16*L && col < CHO*L){
    int i = kk / L,  k2 = kk % L;
    int o = col / L, nn = col % L;
    for (int g = 0; g < 36; ++g){
      float wv = w[((size_t)i*CHO + o)*36 + g];
      float2 d = Dg[(((size_t)g*NL + l)*L + nn)*L + k2];
      ar += wv*d.x; ai += wv*d.y;
    }
    ar *=  scale;
    ai *= -scale;     // conj(Psi)
  }
  Bpr[idx] = (_Float16)ar;
  Bpi[idx] = (_Float16)ai;
}

// ---------------------------------------------------------------------------
// C/D tile store: m = r + 8*(lane>=16), n = lane&15 (ISA 7.12.2 C/D layout)
// ---------------------------------------------------------------------------
__device__ inline void store_tile(const v8f& zrv, const v8f& ziv,
                                  float* __restrict__ Zr, float* __restrict__ Zi,
                                  int mt, int nt, int L, int NL, int CHO, int l, int lane){
  const int Mrows = 8*L, Ncols = CHO*L;
  const int col = nt*16 + (lane & 15);
  if (col >= Ncols) return;
  int o = col / L, nn = col % L;
#pragma unroll
  for (int r = 0; r < 8; ++r){
    int row = mt*16 + r + ((lane >> 4) << 3);
    if (row < Mrows){
      int b = row / L, m = row % L;
      size_t zoff = ((((size_t)b*CHO + o)*NL + l)*L + m)*L + nn;
      Zr[zoff] = zrv[r];
      Zi[zoff] = ziv[r];
    }
  }
}

// ---------------------------------------------------------------------------
// Complex GEMM on pre-packed fragments, 2x2 register-blocked:
//   Z[b,o,l,m,n] = sum_{i,k} Fh[b,i,l,m,k] * PsiC[i,o,l,n,k]
// Each wave owns a 32x32 output patch (2 A-tiles x 2 B-tiles). Per 32-wide
// K chunk: 8 contiguous 32B fragment loads feed 16 WMMAs (2x the FLOP/byte of
// the unblocked form). Edge tiles are pointer-clamped and store-suppressed.
// ---------------------------------------------------------------------------
__global__ void __launch_bounds__(32)
k_zgemm(const _Float16* __restrict__ Apr, const _Float16* __restrict__ Api,
        const _Float16* __restrict__ Bpr, const _Float16* __restrict__ Bpi,
        float* __restrict__ Zr, float* __restrict__ Zi,
        int L, int NL, int CHO, int MT, int NT, int NC){
  const int l   = blockIdx.z;
  const int mt0 = blockIdx.y * 2;
  const int nt0 = blockIdx.x * 2;
  const bool hasM1 = (mt0 + 1) < MT;
  const bool hasN1 = (nt0 + 1) < NT;
  const int mt1 = hasM1 ? mt0 + 1 : mt0;
  const int nt1 = hasN1 ? nt0 + 1 : nt0;
  const int lane = threadIdx.x;
  const size_t laneOff = (size_t)lane << 4;          // 16 halves = 32B

  const _Float16* a0r = Apr + (((size_t)(l*MT + mt0)*NC) << 9) + laneOff;
  const _Float16* a0i = Api + (((size_t)(l*MT + mt0)*NC) << 9) + laneOff;
  const _Float16* a1r = Apr + (((size_t)(l*MT + mt1)*NC) << 9) + laneOff;
  const _Float16* a1i = Api + (((size_t)(l*MT + mt1)*NC) << 9) + laneOff;
  const _Float16* b0r = Bpr + (((size_t)(l*NT + nt0)*NC) << 9) + laneOff;
  const _Float16* b0i = Bpi + (((size_t)(l*NT + nt0)*NC) << 9) + laneOff;
  const _Float16* b1r = Bpr + (((size_t)(l*NT + nt1)*NC) << 9) + laneOff;
  const _Float16* b1i = Bpi + (((size_t)(l*NT + nt1)*NC) << 9) + laneOff;

  v8f zr00 = {}, zr01 = {}, zr10 = {}, zr11 = {};
  v8f zi00 = {}, zi01 = {}, zi10 = {}, zi11 = {};

  for (int ch = 0; ch < NC; ++ch){
    v16h A0r = *(const v16h*)a0r;  v16h A0i = *(const v16h*)a0i;
    v16h A1r = *(const v16h*)a1r;  v16h A1i = *(const v16h*)a1i;
    v16h B0r = *(const v16h*)b0r;  v16h B0i = *(const v16h*)b0i;
    v16h B1r = *(const v16h*)b1r;  v16h B1i = *(const v16h*)b1i;
    __builtin_prefetch((const void*)(a0r + 512), 0, 1);
    __builtin_prefetch((const void*)(a1r + 512), 0, 1);
    __builtin_prefetch((const void*)(b0r + 512), 0, 1);
    __builtin_prefetch((const void*)(b1r + 512), 0, 1);
    a0r += 512; a0i += 512; a1r += 512; a1i += 512;
    b0r += 512; b0i += 512; b1r += 512; b1i += 512;
    v16h N0 = -A0i;
    v16h N1 = -A1i;
    // Zr += Ar*Br - Ai*Bi ; Zi += Ar*Bi + Ai*Br   (for each of the 4 tiles)
    zr00 = __builtin_amdgcn_wmma_f32_16x16x32_f16(false, A0r, false, B0r, (short)0, zr00, false, false);
    zr01 = __builtin_amdgcn_wmma_f32_16x16x32_f16(false, A0r, false, B1r, (short)0, zr01, false, false);
    zr10 = __builtin_amdgcn_wmma_f32_16x16x32_f16(false, A1r, false, B0r, (short)0, zr10, false, false);
    zr11 = __builtin_amdgcn_wmma_f32_16x16x32_f16(false, A1r, false, B1r, (short)0, zr11, false, false);
    zr00 = __builtin_amdgcn_wmma_f32_16x16x32_f16(false, N0,  false, B0i, (short)0, zr00, false, false);
    zr01 = __builtin_amdgcn_wmma_f32_16x16x32_f16(false, N0,  false, B1i, (short)0, zr01, false, false);
    zr10 = __builtin_amdgcn_wmma_f32_16x16x32_f16(false, N1,  false, B0i, (short)0, zr10, false, false);
    zr11 = __builtin_amdgcn_wmma_f32_16x16x32_f16(false, N1,  false, B1i, (short)0, zr11, false, false);
    zi00 = __builtin_amdgcn_wmma_f32_16x16x32_f16(false, A0r, false, B0i, (short)0, zi00, false, false);
    zi01 = __builtin_amdgcn_wmma_f32_16x16x32_f16(false, A0r, false, B1i, (short)0, zi01, false, false);
    zi10 = __builtin_amdgcn_wmma_f32_16x16x32_f16(false, A1r, false, B0i, (short)0, zi10, false, false);
    zi11 = __builtin_amdgcn_wmma_f32_16x16x32_f16(false, A1r, false, B1i, (short)0, zi11, false, false);
    zi00 = __builtin_amdgcn_wmma_f32_16x16x32_f16(false, A0i, false, B0r, (short)0, zi00, false, false);
    zi01 = __builtin_amdgcn_wmma_f32_16x16x32_f16(false, A0i, false, B1r, (short)0, zi01, false, false);
    zi10 = __builtin_amdgcn_wmma_f32_16x16x32_f16(false, A1i, false, B0r, (short)0, zi10, false, false);
    zi11 = __builtin_amdgcn_wmma_f32_16x16x32_f16(false, A1i, false, B1r, (short)0, zi11, false, false);
  }

  store_tile(zr00, zi00, Zr, Zi, mt0, nt0, L, NL, CHO, l, lane);
  if (hasN1)          store_tile(zr01, zi01, Zr, Zi, mt0, nt1, L, NL, CHO, l, lane);
  if (hasM1)          store_tile(zr10, zi10, Zr, Zi, mt1, nt0, L, NL, CHO, l, lane);
  if (hasM1 && hasN1) store_tile(zr11, zi11, Zr, Zi, mt1, nt1, L, NL, CHO, l, lane);
}

// ---------------------------------------------------------------------------
// G = sum_l Wi*Z, inverse DFT back to (u,v), + bias, with mode:
//   0: out = relu(y + bias)        1: out = y + bias        2: out = relu(out + y + bias)
// One (b,o,k) plane per block.
// ---------------------------------------------------------------------------
__global__ void k_ifft(const float* __restrict__ Zr, const float* __restrict__ Zi,
                       const float* __restrict__ Wi, const float* __restrict__ bias,
                       float* out, int L, int NL, int CHO, int NB2, int mode){
  __shared__ float2 G[961];
  __shared__ float2 T[992];
  const int tid = threadIdx.x;
  const int plane = blockIdx.x;
  const int k = plane % NB2;
  const int o = (plane / NB2) % CHO;
  const int b = plane / (NB2 * CHO);
  const int half = (L-1)/2;

  for (int t = tid; t < L*L; t += blockDim.x){
    int p = t / L, q = t % L;
    float gr = 0.f, gi = 0.f;
    for (int l = 0; l < NL; ++l){
      float wv = Wi[(((size_t)k*NL + l)*L + p)*L + q];
      size_t zo = ((((size_t)b*CHO + o)*NL + l)*L + p)*L + q;
      gr += wv*Zr[zo]; gi += wv*Zi[zo];
    }
    G[t] = make_float2(gr, gi);
  }
  __syncthreads();
  for (int t = tid; t < L*NB2; t += blockDim.x){
    int p = t / NB2, v = t % NB2;
    float sr = 0.f, si = 0.f;
    for (int q = 0; q < L; ++q){
      int nf = q - half;
      int ph = ((nf*v) % NB2 + NB2) % NB2;
      float ang = (2.0f*PI_F/(float)NB2) * (float)ph;
      float s, c; __sincosf(ang, &s, &c);
      float2 g = G[p*L + q];
      sr += g.x*c - g.y*s;
      si += g.x*s + g.y*c;
    }
    T[p*NB2 + v] = make_float2(sr, si);
  }
  __syncthreads();
  const float bv = bias[o];
  float* dst = out + (size_t)plane * NB2 * NB2;
  for (int t = tid; t < NB2*NB2; t += blockDim.x){
    int u = t / NB2, v = t % NB2;
    float acc = 0.f;
    for (int p = 0; p < L; ++p){
      int mf = p - half;
      int ph = ((mf*u) % NB2 + NB2) % NB2;
      float ang = (2.0f*PI_F/(float)NB2) * (float)ph;
      float s, c; __sincosf(ang, &s, &c);
      float2 tv = T[p*NB2 + v];
      acc += tv.x*c - tv.y*s;     // real part only
    }
    float val = acc + bv;
    if (mode == 0)      val = fmaxf(val, 0.f);
    else if (mode == 2) val = fmaxf(dst[t] + val, 0.f);
    dst[t] = val;
  }
}

// ---------------------------------------------------------------------------
// Host launcher
// ---------------------------------------------------------------------------
extern "C" void kernel_launch(void* const* d_in, const int* in_sizes, int n_in,
                              void* d_out, int out_size, void* d_ws, size_t ws_size,
                              hipStream_t stream) {
  (void)in_sizes; (void)n_in; (void)out_size; (void)ws_size;
  const float* x   = (const float*)d_in[0];
  const float* w1  = (const float*)d_in[1];
  const float* b1  = (const float*)d_in[2];
  const float* w2  = (const float*)d_in[3];
  const float* b2  = (const float*)d_in[4];
  const float* wsk = (const float*)d_in[5];
  const float* bsk = (const float*)d_in[6];
  float* outp = (float*)d_out;

  // ---- geometry ----
  const int L1 = 31, NL1 = 16, CO1 = 16, NBo1 = 32;   // conv1: b 16 -> 16
  const int L2 = 15, NL2 = 8,  CO2 = 32, NBo2 = 16;   // down : b 16 -> 8
  const int MT1 = (8*L1 + 15)/16,  NT1 = (CO1*L1 + 15)/16,  NC1 = (16*L1 + 31)/32; // 16,31,16
  const int MT2 = (8*L2 + 15)/16,  NT2 = (CO2*L2 + 15)/16,  NC2 = (16*L2 + 31)/32; // 8,30,8
  const float scale1 = 1.0f / 24.0f;                  // 1/sqrt(36*16)
  const float scale2 = 1.0f / sqrtf(72.0f);           // 1/sqrt(36*16/8)

  // ---- workspace carve-up ----
  char* base = (char*)d_ws;
  size_t off = 0;
  auto alloc = [&](size_t bytes) -> char* {
    char* p = base + off;
    off = (off + bytes + 255) & ~(size_t)255;
    return p;
  };
  float*    Wf1 = (float*)   alloc((size_t)32*NL1*L1*L1*4);
  float*    Wi1 = (float*)   alloc((size_t)32*NL1*L1*L1*4);
  float*    Wf2 = (float*)   alloc((size_t)32*NL2*L2*L2*4);
  float*    Wi2 = (float*)   alloc((size_t)16*NL2*L2*L2*4);
  float2*   Dg1 = (float2*)  alloc((size_t)36*NL1*L1*L1*8);
  float2*   Dg2 = (float2*)  alloc((size_t)36*NL2*L2*L2*8);
  _Float16* B1r = (_Float16*)alloc((size_t)NL1*NT1*NC1*512*2);
  _Float16* B1i = (_Float16*)alloc((size_t)NL1*NT1*NC1*512*2);
  _Float16* B2r = (_Float16*)alloc((size_t)NL2*NT2*NC2*512*2);
  _Float16* B2i = (_Float16*)alloc((size_t)NL2*NT2*NC2*512*2);
  _Float16* BSr = (_Float16*)alloc((size_t)NL2*NT2*NC2*512*2);
  _Float16* BSi = (_Float16*)alloc((size_t)NL2*NT2*NC2*512*2);
  float2*   Fzx = (float2*)  alloc((size_t)8*16*32*L1*L1*8);
  _Float16* A1r = (_Float16*)alloc((size_t)NL1*MT1*NC1*512*2);
  _Float16* A1i = (_Float16*)alloc((size_t)NL1*MT1*NC1*512*2);
  _Float16* ASr = (_Float16*)alloc((size_t)NL2*MT2*NC2*512*2);
  _Float16* ASi = (_Float16*)alloc((size_t)NL2*MT2*NC2*512*2);
  float*    out1= (float*)   alloc((size_t)8*16*32*32*32*4);
  float2*   Fzo = (float2*)  alloc((size_t)8*16*32*L2*L2*8);
  _Float16* A2r = (_Float16*)alloc((size_t)NL2*MT2*NC2*512*2);
  _Float16* A2i = (_Float16*)alloc((size_t)NL2*MT2*NC2*512*2);
  float*    Z1r = (float*)   alloc((size_t)8*CO1*NL1*L1*L1*4);
  float*    Z1i = (float*)   alloc((size_t)8*CO1*NL1*L1*L1*4);
  float*    Z2r = (float*)   alloc((size_t)8*CO2*NL2*L2*L2*4);
  float*    Z2i = (float*)   alloc((size_t)8*CO2*NL2*L2*L2*4);
  float*    Zsr = (float*)   alloc((size_t)8*CO2*NL2*L2*L2*4);
  float*    Zsi = (float*)   alloc((size_t)8*CO2*NL2*L2*L2*4);

  auto blocks = [](long tot){ return (unsigned)((tot + 255) / 256); };

  // ---- constants ----
  k_wig<<<blocks((long)32*NL1*L1*L1), 256, 0, stream>>>(Wf1, 32, NL1, L1, PI_D/64.0, 0);
  k_wig<<<blocks((long)32*NL1*L1*L1), 256, 0, stream>>>(Wi1, 32, NL1, L1, PI_D/64.0, 1);
  k_wig<<<blocks((long)32*NL2*L2*L2), 256, 0, stream>>>(Wf2, 32, NL2, L2, PI_D/64.0, 0);
  k_wig<<<blocks((long)16*NL2*L2*L2), 256, 0, stream>>>(Wi2, 16, NL2, L2, PI_D/32.0, 1);
  k_dg <<<blocks(36L*NL1*L1*L1), 256, 0, stream>>>(Dg1, NL1, L1);
  k_dg <<<blocks(36L*NL2*L2*L2), 256, 0, stream>>>(Dg2, NL2, L2);

  // ---- Psi -> packed B fragments ----
  k_psi_pack<<<blocks((long)NL1*NT1*NC1*512), 256, 0, stream>>>(w1,  Dg1, B1r, B1i, CO1, NL1, L1, NT1, NC1, scale1);
  k_psi_pack<<<blocks((long)NL2*NT2*NC2*512), 256, 0, stream>>>(w2,  Dg2, B2r, B2i, CO2, NL2, L2, NT2, NC2, scale2);
  k_psi_pack<<<blocks((long)NL2*NT2*NC2*512), 256, 0, stream>>>(wsk, Dg2, BSr, BSi, CO2, NL2, L2, NT2, NC2, scale2);

  // ---- conv1 + skip front end: DFT of x, beta contraction -> packed A ----
  k_dft<<<8*16*32, 256, 0, stream>>>(x, Fzx, L1);
  k_fh_pack<<<blocks((long)NL1*MT1*NC1*512), 256, 0, stream>>>(Fzx, L1, 0, Wf1, A1r, A1i, NL1, L1, MT1, NC1);
  k_fh_pack<<<blocks((long)NL2*MT2*NC2*512), 256, 0, stream>>>(Fzx, L1, 8, Wf2, ASr, ASi, NL2, L2, MT2, NC2);

  // ---- conv1 Z-einsum (WMMA, 2x2 blocked) + back end ----
  {
    dim3 g((NT1 + 1)/2, (MT1 + 1)/2, NL1);
    k_zgemm<<<g, 32, 0, stream>>>(A1r, A1i, B1r, B1i, Z1r, Z1i, L1, NL1, CO1, MT1, NT1, NC1);
  }
  k_ifft<<<8*CO1*NBo1, 256, 0, stream>>>(Z1r, Z1i, Wi1, b1, out1, L1, NL1, CO1, NBo1, 0);

  // ---- conv2 front end on out1 ----
  k_dft<<<8*16*32, 256, 0, stream>>>(out1, Fzo, L2);
  k_fh_pack<<<blocks((long)NL2*MT2*NC2*512), 256, 0, stream>>>(Fzo, L2, 0, Wf2, A2r, A2i, NL2, L2, MT2, NC2);

  // ---- down-conv Z-einsums (WMMA, 2x2 blocked) ----
  {
    dim3 g((NT2 + 1)/2, (MT2 + 1)/2, NL2);
    k_zgemm<<<g, 32, 0, stream>>>(A2r, A2i, B2r, B2i, Z2r, Z2i, L2, NL2, CO2, MT2, NT2, NC2);
    k_zgemm<<<g, 32, 0, stream>>>(ASr, ASi, BSr, BSi, Zsr, Zsi, L2, NL2, CO2, MT2, NT2, NC2);
  }

  // ---- back ends: conv2 writes, skip accumulates + final relu ----
  k_ifft<<<8*CO2*NBo2, 256, 0, stream>>>(Z2r, Z2i, Wi2, b2,  outp, L2, NL2, CO2, NBo2, 1);
  k_ifft<<<8*CO2*NBo2, 256, 0, stream>>>(Zsr, Zsi, Wi2, bsk, outp, L2, NL2, CO2, NBo2, 2);
}
